// MDN_module_21655225106918
// MI455X (gfx1250) — matrix operand order repaired and checked by reading, not compiled, not run on hardware
//
#include <hip/hip_runtime.h>
#include <hip/hip_bf16.h>

// ---------------------------------------------------------------------------
// MDN module on MI455X (gfx1250, wave32, WMMA).
// - All four GEMMs via v_wmma_f32_16x16x32_bf16 (f32 accumulate).
// - Weights converted+transposed to bf16 once (B fragments contiguous).
// - A row-tiles staged to LDS once per block via async global->LDS copy
//   (gfx1250 ASYNCcnt path) and read back as ds_load_b128 fragments.
// ---------------------------------------------------------------------------

#define B_DIM 8192
#define N_DIM 256
#define H_DIM 2048
#define F_DIM 512          // 2*N
#define BETA_F 0.99f
#define LOG2PI_F 1.8378770664093453f

#ifndef __has_builtin
#define __has_builtin(x) 0
#endif
#if __has_builtin(__builtin_amdgcn_global_load_async_to_lds_b128) && \
    __has_builtin(__builtin_amdgcn_s_wait_asynccnt)
#define USE_ASYNC_LDS 1
#else
#define USE_ASYNC_LDS 0
#endif

typedef __attribute__((ext_vector_type(16))) __bf16 v16bf;
typedef __attribute__((ext_vector_type(8)))  __bf16 v8bf;
typedef __attribute__((ext_vector_type(8)))  float  v8f;
typedef __attribute__((ext_vector_type(4)))  int    v4i;

#if USE_ASYNC_LDS
typedef __attribute__((address_space(1))) v4i* gv4i_p;   // global v4i*
typedef __attribute__((address_space(3))) v4i* lv4i_p;   // LDS v4i*
#endif

#define WMMA_BF16(a, b, c) \
    __builtin_amdgcn_wmma_f32_16x16x32_bf16(false, (a), false, (b), (short)0, (c), false, false)

// ---- WMMA fragment loaders (contiguous k = 16*(L>>4) + s, same order for A and B) ------------------
// A (16x32 bf16, row-major): lane L holds row m = L&15;
//   VGPR0-3 = K (L>>4)*16 .. +7 ; VGPR4-7 = +8..+15
__device__ __forceinline__ v16bf load_a_frag(const __bf16* __restrict__ base,
                                             int ld, int k, int lane) {
    const int m  = lane & 15;
    const int kh = (lane >> 4) << 4;      // 0 or 16
    const __bf16* p = base + (size_t)m * ld + (k + kh);
    v8bf lo = *(const v8bf*)(p);          // K = kh .. kh+7
    v8bf hi = *(const v8bf*)(p + 8);      // K = kh+8 .. kh+15
    return __builtin_shufflevector(lo, hi, 0, 1, 2, 3, 4, 5, 6, 7,
                                           8, 9, 10, 11, 12, 13, 14, 15);
}

// B (32x16 bf16) from pre-transposed weight WT[col][k]:
//   lane L holds column n = L&15; lanes 0-15: K=0..15, lanes 16-31: K=16..31
__device__ __forceinline__ v16bf load_b_frag(const __bf16* __restrict__ baseT,
                                             int ld, int col0, int k, int lane) {
    const int n  = lane & 15;
    const int kh = (lane >> 4) << 4;      // 0 or 16
    const __bf16* p = baseT + (size_t)(col0 + n) * ld + (k + kh);
    return *(const v16bf*)(p);            // 16 contiguous bf16 (32B)
}

// ---- stage a contiguous global region into LDS (async path if available) ---
// n_chunks16 = bytes/16; called by all 256 threads of the block.
__device__ __forceinline__ void stage_to_lds(__bf16* dst_lds,
                                             const __bf16* src_g,
                                             int n_chunks16) {
    for (int c = threadIdx.x; c < n_chunks16; c += 256) {
#if USE_ASYNC_LDS
        __builtin_amdgcn_global_load_async_to_lds_b128(
            (gv4i_p)(src_g + (size_t)c * 8),
            (lv4i_p)(dst_lds + (size_t)c * 8),
            0, 0);
#else
        ((int4*)dst_lds)[c] = ((const int4*)src_g)[c];
#endif
    }
#if USE_ASYNC_LDS
    __builtin_amdgcn_s_wait_asynccnt(0);
#endif
    __syncthreads();
}

// ---- conversion kernels ----------------------------------------------------
__global__ void k_cvt_bf16(const float* __restrict__ in, __bf16* __restrict__ out, int n) {
    int i = blockIdx.x * blockDim.x + threadIdx.x;
    if (i < n) out[i] = (__bf16)in[i];
}

// in: [R,C] f32 row-major -> out: [C,R] bf16 row-major (transpose + convert)
__global__ void k_tcvt_bf16(const float* __restrict__ in, __bf16* __restrict__ out,
                            int R, int C) {
    int i = blockIdx.x * blockDim.x + threadIdx.x;
    if (i >= R * C) return;
    int r = i / C, c = i % C;
    out[(size_t)c * R + r] = (__bf16)in[i];
}

__global__ void k_init(float* __restrict__ Vx, float* __restrict__ Vmu,
                       float* __restrict__ logp_acc) {
    int i = blockIdx.x * blockDim.x + threadIdx.x;
    if (i < B_DIM) { Vx[i] = 1e-3f; Vmu[i] = 1e-3f; }
    if (i == 0) *logp_acc = 0.0f;
}

// ---- GEMM1: h = tanh(x @ W1 + b1)  [8192 x 2048], K=256 --------------------
// grid = (H/512, B/16); block = 8 waves; A row-tile (16x256, 8 KB) in LDS.
__global__ void __launch_bounds__(256) k_gemm1(const __bf16* __restrict__ A,
                                               const __bf16* __restrict__ W1T,
                                               const float*  __restrict__ b1,
                                               __bf16* __restrict__ H) {
    __shared__ __attribute__((aligned(32))) __bf16 sA[16 * N_DIM];   // 8 KB
    const int lane = threadIdx.x & 31;
    const int wav  = threadIdx.x >> 5;
    const int rm   = blockIdx.y * 16;
    const int cn   = blockIdx.x * 512 + wav * 64;

    stage_to_lds(sA, A + (size_t)rm * N_DIM, 16 * N_DIM * 2 / 16);

    v8f acc[4];
#pragma unroll
    for (int g = 0; g < 4; ++g)
        acc[g] = (v8f){0.f, 0.f, 0.f, 0.f, 0.f, 0.f, 0.f, 0.f};

    for (int k = 0; k < N_DIM; k += 32) {
        v16bf a = load_a_frag(sA, N_DIM, k, lane);         // ds_load path
#pragma unroll
        for (int g = 0; g < 4; ++g) {
            v16bf b = load_b_frag(W1T, N_DIM, cn + 16 * g, k, lane);
            acc[g] = WMMA_BF16(a, b, acc[g]);
        }
    }

    const int n  = lane & 15;
    const int mh = (lane >> 4) << 3;
#pragma unroll
    for (int g = 0; g < 4; ++g) {
        const int col = cn + 16 * g + n;
        const float bv = b1[col];
#pragma unroll
        for (int v = 0; v < 8; ++v) {
            const int row = rm + mh + v;
            H[(size_t)row * H_DIM + col] = (__bf16)tanhf(acc[g][v] + bv);
        }
    }
}

// ---- GEMM2: f = h @ W2 + b2 -> split mu (f32+bf16) / var=exp(logvar) -------
// grid = B/16; block = 8 waves covering all 512 cols; A tile (16x2048, 64 KB).
__global__ void __launch_bounds__(256) k_gemm2(const __bf16* __restrict__ Hb,
                                               const __bf16* __restrict__ W2T,
                                               const float*  __restrict__ b2,
                                               float*  __restrict__ mu,
                                               __bf16* __restrict__ mubf,
                                               float*  __restrict__ var) {
    __shared__ __attribute__((aligned(32))) __bf16 sA[16 * H_DIM];   // 64 KB
    const int lane = threadIdx.x & 31;
    const int wav  = threadIdx.x >> 5;
    const int rm   = blockIdx.x * 16;
    const int cn   = wav * 64;

    stage_to_lds(sA, Hb + (size_t)rm * H_DIM, 16 * H_DIM * 2 / 16);

    v8f acc[4];
#pragma unroll
    for (int g = 0; g < 4; ++g)
        acc[g] = (v8f){0.f, 0.f, 0.f, 0.f, 0.f, 0.f, 0.f, 0.f};

    for (int k = 0; k < H_DIM; k += 32) {
        v16bf a = load_a_frag(sA, H_DIM, k, lane);
#pragma unroll
        for (int g = 0; g < 4; ++g) {
            v16bf b = load_b_frag(W2T, H_DIM, cn + 16 * g, k, lane);
            acc[g] = WMMA_BF16(a, b, acc[g]);
        }
    }

    const int n  = lane & 15;
    const int mh = (lane >> 4) << 3;
#pragma unroll
    for (int g = 0; g < 4; ++g) {
        const int col = cn + 16 * g + n;              // uniform side per g
        const float bv = b2[col];
        if (cn + 16 * g < N_DIM) {                    // mu half
#pragma unroll
            for (int v = 0; v < 8; ++v) {
                const int row = rm + mh + v;
                const float f = acc[g][v] + bv;
                mu  [(size_t)row * N_DIM + col] = f;
                mubf[(size_t)row * N_DIM + col] = (__bf16)f;
            }
        } else {                                      // logvar half
#pragma unroll
            for (int v = 0; v < 8; ++v) {
                const int row = rm + mh + v;
                var[(size_t)row * N_DIM + (col - N_DIM)] = expf(acc[g][v] + bv);
            }
        }
    }
}

// ---- V(z) = rowsum((z @ Wv)^2) accumulated atomically (V pre-init = 1e-3) --
// grid = (N/128, B/16); block = 8 waves, one 16x16 u-tile each; A tile 8 KB.
__global__ void __launch_bounds__(256) k_vquad(const __bf16* __restrict__ A,
                                               const __bf16* __restrict__ WvT,
                                               float* __restrict__ Vout) {
    __shared__ __attribute__((aligned(32))) __bf16 sA[16 * N_DIM];   // 8 KB
    const int lane = threadIdx.x & 31;
    const int wav  = threadIdx.x >> 5;
    const int rm   = blockIdx.y * 16;
    const int cn   = (blockIdx.x * 8 + wav) * 16;

    stage_to_lds(sA, A + (size_t)rm * N_DIM, 16 * N_DIM * 2 / 16);

    v8f c = (v8f){0.f, 0.f, 0.f, 0.f, 0.f, 0.f, 0.f, 0.f};
    for (int k = 0; k < N_DIM; k += 32) {
        v16bf a = load_a_frag(sA, N_DIM, k, lane);
        v16bf b = load_b_frag(WvT, N_DIM, cn, k, lane);
        c = WMMA_BF16(a, b, c);
    }

    const int mh = (lane >> 4) << 3;
#pragma unroll
    for (int v = 0; v < 8; ++v) {
        float s = c[v] * c[v];
        s += __shfl_xor(s, 1, 32);                    // reduce over 16 col lanes
        s += __shfl_xor(s, 2, 32);
        s += __shfl_xor(s, 4, 32);
        s += __shfl_xor(s, 8, 32);
        if ((lane & 15) == 0)
            atomicAdd(&Vout[rm + mh + v], s);
    }
}

// ---- fused epilogue: scale mu, sample fx, diag-MVN logp, scalar reduce -----
__global__ void __launch_bounds__(256) k_final(const float* __restrict__ mu,
                                               const float* __restrict__ var,
                                               const float* __restrict__ Vx,
                                               const float* __restrict__ Vmu,
                                               const float* __restrict__ y,
                                               const float* __restrict__ eps,
                                               float* __restrict__ fx_out,
                                               float* __restrict__ logp_acc) {
    const int i = blockIdx.x;                         // row (sample)
    const int j = threadIdx.x;                        // column (0..255)

    const float vx  = Vx[i];
    const float vmu = Vmu[i];
    const float bvx = BETA_F * vx;
    const float rel = fmaxf(bvx - vmu, 0.0f);
    const float scale = (bvx - rel) / vmu;

    const size_t idx = (size_t)i * N_DIM + j;
    const float ms = mu[idx] * scale;
    const float vr = var[idx];
    fx_out[idx] = ms + sqrtf(vr) * eps[idx];

    const float d = y[idx] - ms;
    float term = logf(vr) + d * d / vr;               // log var + q

    term += __shfl_xor(term, 16, 32);
    term += __shfl_xor(term, 8, 32);
    term += __shfl_xor(term, 4, 32);
    term += __shfl_xor(term, 2, 32);
    term += __shfl_xor(term, 1, 32);

    __shared__ float sred[8];
    const int lane = threadIdx.x & 31;
    const int w    = threadIdx.x >> 5;
    if (lane == 0) sred[w] = term;
    __syncthreads();
    if (threadIdx.x == 0) {
        float tot = 0.0f;
#pragma unroll
        for (int q = 0; q < 8; ++q) tot += sred[q];
        // logp_i = -0.5*(N*log2pi + tot); logp_y = -sum logp_i
        atomicAdd(logp_acc, 0.5f * ((float)N_DIM * LOG2PI_F + tot));
    }
}

// ---------------------------------------------------------------------------
extern "C" void kernel_launch(void* const* d_in, const int* in_sizes, int n_in,
                              void* d_out, int out_size, void* d_ws, size_t ws_size,
                              hipStream_t stream) {
    const float* x   = (const float*)d_in[0];   // [B,N]
    const float* y   = (const float*)d_in[1];   // [B,1,N]
    const float* eps = (const float*)d_in[2];   // [B,1,N]
    const float* W1  = (const float*)d_in[3];   // [N,H]
    const float* b1  = (const float*)d_in[4];   // [H]
    const float* W2  = (const float*)d_in[5];   // [H,2N]
    const float* b2  = (const float*)d_in[6];   // [2N]
    const float* Wv  = (const float*)d_in[7];   // [N,N]

    float* out = (float*)d_out;                 // fx [B*N] then logp_y scalar

    // workspace carve-up (256B aligned slices), ~62 MB total
    char* ws = (char*)d_ws;
    size_t off = 0;
    auto carve = [&](size_t bytes) -> char* {
        char* p = ws + off;
        off = (off + bytes + 255) & ~(size_t)255;
        return p;
    };
    __bf16* xb   = (__bf16*)carve((size_t)B_DIM * N_DIM * 2);   // x in bf16
    __bf16* w1t  = (__bf16*)carve((size_t)H_DIM * N_DIM * 2);   // W1^T bf16
    __bf16* w2t  = (__bf16*)carve((size_t)F_DIM * H_DIM * 2);   // W2^T bf16
    __bf16* wvt  = (__bf16*)carve((size_t)N_DIM * N_DIM * 2);   // Wv^T bf16
    __bf16* hbf  = (__bf16*)carve((size_t)B_DIM * H_DIM * 2);   // tanh layer
    float*  mu   = (float*) carve((size_t)B_DIM * N_DIM * 4);
    __bf16* mubf = (__bf16*)carve((size_t)B_DIM * N_DIM * 2);
    float*  var  = (float*) carve((size_t)B_DIM * N_DIM * 4);
    float*  Vx   = (float*) carve((size_t)B_DIM * 4);
    float*  Vmu  = (float*) carve((size_t)B_DIM * 4);
    (void)ws_size; (void)in_sizes; (void)n_in; (void)out_size;

    // 1) precision conversion / weight transposes
    k_cvt_bf16 <<<(B_DIM * N_DIM) / 256, 256, 0, stream>>>(x, xb, B_DIM * N_DIM);
    k_tcvt_bf16<<<(N_DIM * H_DIM) / 256, 256, 0, stream>>>(W1, w1t, N_DIM, H_DIM);
    k_tcvt_bf16<<<(H_DIM * F_DIM) / 256, 256, 0, stream>>>(W2, w2t, H_DIM, F_DIM);
    k_tcvt_bf16<<<(N_DIM * N_DIM) / 256, 256, 0, stream>>>(Wv, wvt, N_DIM, N_DIM);
    k_init     <<<B_DIM / 256, 256, 0, stream>>>(Vx, Vmu, out + (size_t)B_DIM * N_DIM);

    // 2) h = tanh(x@W1+b1): blocks = (2048/512) x (8192/16)
    k_gemm1<<<dim3(H_DIM / 512, B_DIM / 16), 256, 0, stream>>>(xb, w1t, b1, hbf);

    // 3) Vx = ||x@Wv||^2 + 1e-3 : blocks = (256/128) x (8192/16)
    k_vquad<<<dim3(N_DIM / 128, B_DIM / 16), 256, 0, stream>>>(xb, wvt, Vx);

    // 4) f = h@W2+b2 -> mu / var : one block per 16-row tile
    k_gemm2<<<B_DIM / 16, 256, 0, stream>>>(hbf, w2t, b2, mu, mubf, var);

    // 5) Vmu = ||mu@Wv||^2 + 1e-3
    k_vquad<<<dim3(N_DIM / 128, B_DIM / 16), 256, 0, stream>>>(mubf, wvt, Vmu);

    // 6) fused epilogue + logp reduction
    k_final<<<B_DIM, 256, 0, stream>>>(mu, var, Vx, Vmu, y, eps,
                                       out, out + (size_t)B_DIM * N_DIM);
}
